// VariationalGCNEncoder_22823456211324
// MI455X (gfx1250) — matrix operand order repaired and checked by reading, not compiled, not run on hardware
//
#include <hip/hip_runtime.h>
#include <hip/hip_bf16.h>

typedef __attribute__((ext_vector_type(16))) __bf16 v16bf;
typedef __attribute__((ext_vector_type(8)))  __bf16 v8bf;
typedef __attribute__((ext_vector_type(8)))  float  v8f;

#define C_IN  256
#define C_HID 128

// ---------------- conversion kernels ----------------
__global__ void k_f32_to_bf16(const float* __restrict__ in, __bf16* __restrict__ out, size_t n) {
  size_t i = (size_t)blockIdx.x * blockDim.x + threadIdx.x;
  if (i < n) out[i] = (__bf16)in[i];
}

__global__ void k_relu_to_bf16(const float* __restrict__ in, __bf16* __restrict__ out, size_t n) {
  size_t i = (size_t)blockIdx.x * blockDim.x + threadIdx.x;
  if (i < n) out[i] = (__bf16)fmaxf(in[i], 0.0f);
}

// W: [K, Nc] row-major f32  ->  WT: [Nc, K] row-major bf16
__global__ void k_transpose_to_bf16(const float* __restrict__ W, __bf16* __restrict__ WT,
                                    int K, int Nc) {
  int idx = blockIdx.x * blockDim.x + threadIdx.x;
  if (idx >= K * Nc) return;
  int k = idx / Nc, n = idx % Nc;
  WT[(size_t)n * K + k] = (__bf16)W[idx];
}

// Fuse Wmu [128,64] and Wls [128,64] -> WT [128 cols][128 K] (bf16, transposed)
__global__ void k_wcat_transpose(const float* __restrict__ Wmu, const float* __restrict__ Wls,
                                 __bf16* __restrict__ WT) {
  int idx = blockIdx.x * blockDim.x + threadIdx.x;
  if (idx >= 128 * 128) return;
  int k = idx >> 7, n = idx & 127;
  float v = (n < 64) ? Wmu[k * 64 + n] : Wls[k * 64 + (n - 64)];
  WT[(size_t)n * 128 + k] = (__bf16)v;
}

// ---------------- degree / normalization ----------------
__global__ void k_deg_init(float* __restrict__ deg, int n) {
  int i = blockIdx.x * blockDim.x + threadIdx.x;
  if (i < n) deg[i] = 1.0f;  // self-loop
}

__global__ void k_deg_count(const long long* __restrict__ dst, float* __restrict__ deg, int E) {
  int i = blockIdx.x * blockDim.x + threadIdx.x;
  if (i < E) atomicAdd(&deg[dst[i]], 1.0f);
}

__global__ void k_deg_rsqrt(float* __restrict__ deg, int n) {
  int i = blockIdx.x * blockDim.x + threadIdx.x;
  if (i < n) deg[i] = rsqrtf(deg[i]);  // deg >= 1 always
}

// ---------------- WMMA GEMM: C[M,128] = A[M,K] (bf16) x BT[128,K]^T (bf16) ----------------
// Block = 256 threads = 8 waves. BT (<=64KB) is staged once into LDS (padded rows,
// conflict-free ds_load_b128 fragment reads); each wave computes a 16-row x 128-col band.
template <int K>
__global__ __launch_bounds__(256) void k_wmma_gemm(const __bf16* __restrict__ A,
                                                   const __bf16* __restrict__ BT,
                                                   float* __restrict__ C, int M) {
  constexpr int KP = K + 8;            // padded row stride: bank advance 4/row, 16B aligned
  extern __shared__ __bf16 sB[];       // 128 * KP bf16, sized at launch

  // --- cooperative stage: BT[128][K] -> sB[128][KP] via b128 copies ---
  {
    constexpr int CPR = K / 8;         // 16-byte chunks per row
    for (int c = threadIdx.x; c < 128 * CPR; c += 256) {
      int row = c / CPR;
      int off = (c - row * CPR) * 8;
      *(uint4*)(sB + (size_t)row * KP + off) =
          *(const uint4*)(BT + (size_t)row * K + off);
    }
  }
  __syncthreads();

  const int wave = (int)((blockIdx.x * blockDim.x + threadIdx.x) >> 5);
  const int lane = threadIdx.x & 31;
  const int row0 = wave * 16;
  if (row0 >= M) return;

  const int r    = lane & 15;   // row within A tile / col within B tile
  const int half = lane >> 4;   // K-chunk selector

  // A 16-bit fragment layout (ISA 7.12.2): lane<16 holds K = k+0..7 and k+16..23;
  // lane>=16 holds K = k+8..15 and k+24..31, for row M = lane&15.
  const __bf16* arow = A + (size_t)(row0 + r) * K + half * 8;

  v8f acc[8];
#pragma unroll
  for (int n = 0; n < 8; ++n)
#pragma unroll
    for (int i = 0; i < 8; ++i) acc[n][i] = 0.0f;

  for (int k = 0; k < K; k += 32) {
    __builtin_prefetch(arow + k + 128, 0, 3);  // global_prefetch_b8 on the A stream
    v8bf a0 = *(const v8bf*)(arow + k);
    v8bf a1 = *(const v8bf*)(arow + k + 16);
    v16bf afrag;
#pragma unroll
    for (int i = 0; i < 8; ++i) { afrag[i] = a0[i]; afrag[i + 8] = a1[i]; }

    // Batch all 8 B fragments from LDS (ds_load_b128 x16), then 8 back-to-back WMMAs.
    v16bf bfrag[8];
#pragma unroll
    for (int n = 0; n < 8; ++n) {
      // B 16-bit fragment: lane<16 holds K = k+0..7 and k+16..23, lane>=16 holds K = k+8..15 and k+24..31,
      // for column N = lane&15.
      const __bf16* bp = sB + (size_t)(n * 16 + r) * KP + k + half * 8;
      v8bf b0 = *(const v8bf*)bp;
      v8bf b1 = *(const v8bf*)(bp + 16);
#pragma unroll
      for (int i = 0; i < 8; ++i) { bfrag[n][i] = b0[i]; bfrag[n][i + 8] = b1[i]; }
    }
#pragma unroll
    for (int n = 0; n < 8; ++n) {
      acc[n] = __builtin_amdgcn_wmma_f32_16x16x32_bf16(
          /*neg_a=*/false, afrag, /*neg_b=*/false, bfrag[n],
          /*c_mod=*/(short)0, acc[n], /*reuse_a=*/false, /*reuse_b=*/false);
    }
  }

  // C/D layout: VGPR i -> row (i + 8*half), col = n*16 + (lane&15)
#pragma unroll
  for (int n = 0; n < 8; ++n) {
    const int col = n * 16 + r;
#pragma unroll
    for (int i = 0; i < 8; ++i) {
      const int m = i + 8 * half;
      C[(size_t)(row0 + m) * 128 + col] = acc[n][i];
    }
  }
}

// ---------------- aggregation (layer 1, 128 features) ----------------
// out[n][f] = inv[n]^2 * lin[n][f] + bias[f]   (self-loop + bias)
__global__ void k_agg_init128(const float* __restrict__ lin, const float* __restrict__ inv,
                              const float* __restrict__ bias, float* __restrict__ out, int N) {
  size_t idx = (size_t)blockIdx.x * blockDim.x + threadIdx.x;
  if (idx >= (size_t)N * 128) return;
  int node = (int)(idx >> 7);
  int f = (int)(idx & 127);
  float iv = inv[node];
  out[idx] = iv * iv * lin[idx] + bias[f];
}

// block = one edge, 128 threads = 128 features; src/dst loads are block-uniform (scalarized)
__global__ void k_edge_scatter128(const long long* __restrict__ src, const long long* __restrict__ dst,
                                  const float* __restrict__ inv, const float* __restrict__ lin,
                                  float* __restrict__ out, int E) {
  int e = blockIdx.x;
  if (e >= E) return;
  int f = threadIdx.x;
  long long s = src[e], d = dst[e];
  float w = inv[s] * inv[d];
  atomicAdd(&out[(size_t)d * 128 + f], w * lin[(size_t)s * 128 + f]);
}

// ---------------- aggregation (layers 2/3 fused; write straight into d_out) ----------------
// lin[n][f]: f<64 -> mu channel f, f>=64 -> logstd channel f-64.
__global__ void k_agg_init_out(const float* __restrict__ lin, const float* __restrict__ inv,
                               const float* __restrict__ bmu, const float* __restrict__ bls,
                               float* __restrict__ out, int N) {
  size_t idx = (size_t)blockIdx.x * blockDim.x + threadIdx.x;
  if (idx >= (size_t)N * 128) return;
  int node = (int)(idx >> 7);
  int f = (int)(idx & 127);
  float iv = inv[node];
  float v = iv * iv * lin[idx];
  if (f < 64) out[(size_t)node * 64 + f] = v + bmu[f];
  else        out[(size_t)N * 64 + (size_t)node * 64 + (f - 64)] = v + bls[f - 64];
}

__global__ void k_edge_scatter_out(const long long* __restrict__ src, const long long* __restrict__ dst,
                                   const float* __restrict__ inv, const float* __restrict__ lin,
                                   float* __restrict__ out, int E, int N) {
  int e = blockIdx.x;
  if (e >= E) return;
  int f = threadIdx.x;
  long long s = src[e], d = dst[e];
  float w = inv[s] * inv[d];
  float v = w * lin[(size_t)s * 128 + f];
  if (f < 64) atomicAdd(&out[(size_t)d * 64 + f], v);
  else        atomicAdd(&out[(size_t)N * 64 + (size_t)d * 64 + (f - 64)], v);
}

// ---------------- host launcher ----------------
extern "C" void kernel_launch(void* const* d_in, const int* in_sizes, int n_in,
                              void* d_out, int out_size, void* d_ws, size_t ws_size,
                              hipStream_t stream) {
  const float*     x   = (const float*)d_in[0];
  const long long* ei  = (const long long*)d_in[1];   // int64 edge_index [2, E]
  const float*     W1  = (const float*)d_in[2];
  const float*     b1  = (const float*)d_in[3];
  const float*     Wmu = (const float*)d_in[4];
  const float*     bmu = (const float*)d_in[5];
  const float*     Wls = (const float*)d_in[6];
  const float*     bls = (const float*)d_in[7];
  float* out = (float*)d_out;

  const int N = in_sizes[0] / C_IN;     // 100000
  const int E = in_sizes[1] / 2;        // 1600000
  const long long* src = ei;
  const long long* dst = ei + E;

  // workspace bump allocator (256B-aligned)
  char* p = (char*)d_ws;
  auto alloc = [&](size_t bytes) -> void* {
    void* q = (void*)p;
    p += (bytes + 255) & ~(size_t)255;
    return q;
  };
  __bf16* x_bf  = (__bf16*)alloc((size_t)N * C_IN * sizeof(__bf16));    // 51.2 MB
  __bf16* W1T   = (__bf16*)alloc((size_t)C_HID * C_IN * sizeof(__bf16));
  __bf16* WcatT = (__bf16*)alloc((size_t)128 * 128 * sizeof(__bf16));
  float*  XW    = (float*) alloc((size_t)N * C_HID * sizeof(float));    // 51.2 MB
  float*  H1    = (float*) alloc((size_t)N * C_HID * sizeof(float));    // 51.2 MB
  __bf16* H_bf  = (__bf16*)alloc((size_t)N * C_HID * sizeof(__bf16));   // 25.6 MB
  float*  L2    = (float*) alloc((size_t)N * 128 * sizeof(float));      // 51.2 MB
  float*  inv   = (float*) alloc((size_t)N * sizeof(float));

  // --- degree -> inv_sqrt (shared by all three conv layers) ---
  k_deg_init <<<(N + 255) / 256, 256, 0, stream>>>(inv, N);
  k_deg_count<<<(E + 255) / 256, 256, 0, stream>>>(dst, inv, E);
  k_deg_rsqrt<<<(N + 255) / 256, 256, 0, stream>>>(inv, N);

  // --- bf16 conversions ---
  size_t nx = (size_t)N * C_IN;
  k_f32_to_bf16<<<(int)((nx + 255) / 256), 256, 0, stream>>>(x, x_bf, nx);
  k_transpose_to_bf16<<<(C_IN * C_HID + 255) / 256, 256, 0, stream>>>(W1, W1T, C_IN, C_HID);
  k_wcat_transpose<<<(128 * 128 + 255) / 256, 256, 0, stream>>>(Wmu, Wls, WcatT);

  // --- layer 1: XW = x @ W1 via WMMA (B staged in LDS) ---
  int waves  = (N + 15) / 16;                 // N = 100000 -> 6250 full tiles
  int blocks = (waves * 32 + 255) / 256;
  size_t sh1 = (size_t)128 * (C_IN + 8) * sizeof(__bf16);    // 67.6 KB
  k_wmma_gemm<C_IN><<<blocks, 256, sh1, stream>>>(x_bf, W1T, XW, N);

  // --- layer 1 aggregation + ReLU ---
  size_t nh = (size_t)N * C_HID;
  k_agg_init128    <<<(int)((nh + 255) / 256), 256, 0, stream>>>(XW, inv, b1, H1, N);
  k_edge_scatter128<<<E, 128, 0, stream>>>(src, dst, inv, XW, H1, E);
  k_relu_to_bf16   <<<(int)((nh + 255) / 256), 256, 0, stream>>>(H1, H_bf, nh);

  // --- layers 2/3 fused: L2 = H @ [Wmu | Wls] via WMMA ---
  size_t sh2 = (size_t)128 * (C_HID + 8) * sizeof(__bf16);   // 34.8 KB
  k_wmma_gemm<C_HID><<<blocks, 256, sh2, stream>>>(H_bf, WcatT, L2, N);

  // --- final aggregation straight into d_out (mu | logstd) ---
  k_agg_init_out    <<<(int)((nh + 255) / 256), 256, 0, stream>>>(L2, inv, bmu, bls, out, N);
  k_edge_scatter_out<<<E, 128, 0, stream>>>(src, dst, inv, L2, out, E, N);
}
